// MATN_11261404250210
// MI455X (gfx1250) — matrix operand order repaired and checked by reading, not compiled, not run on hardware
//
#include <hip/hip_runtime.h>
#include <hip/hip_bf16.h>
#include <math.h>

typedef __attribute__((ext_vector_type(16))) _Float16 v16h;
typedef __attribute__((ext_vector_type(8)))  _Float16 v8h;
typedef __attribute__((ext_vector_type(8)))  float    v8f;

#define D_EMB 256
#define H_DIM 128
#define TILE_B 64
#define LDU 264   // 256 + 8-half pad -> row stride 528B = 132 dwords = +4 banks/row
#define LDH 136   // 128 + 8-half pad -> row stride 272B =  68 dwords = +4 banks/row

// ---------------------------------------------------------------------------
// One GEMM stage: out[64 x N-slice] = act(A[64 x KTOT] @ Wt^T + bias)
// Wave owns one 16-wide N tile at n0; loops 4 M-tiles x K-chunks of 32.
// WMMA fragment layouts per CDNA5 ISA 7.12.2 (wave32):
//   A 16x32 f16 : lane(m=l&15, hi=l>>4): halves 0..7 = K k0+hi*8..+7,
//                 halves 8..15 = K k0+16+hi*8..+7  (two 16B LDS loads)
//   B 32x16 f16 : lane(n=l&15, hi=l>>4): halves 0..7 = K k0+hi*8..+7,
//                 halves 8..15 = K k0+16+hi*8..+7 (two 16B loads from transposed f16 weights)
//   C/D 16x16 f32: vgpr r, lane l -> row r+8*(l>=16), col l&15
// LDS rows padded by 8 halves so the 16 row-strided b128 A-loads hit
// disjoint bank groups (stride = 132 or 68 dwords == +4 banks per row).
// ---------------------------------------------------------------------------
template<int KTOT, int LDA, bool RELU>
__device__ __forceinline__ void gemm_stage(const _Float16* __restrict__ A_lds,
                                           const _Float16* __restrict__ Wt,
                                           int n0,
                                           const float* __restrict__ bias,
                                           _Float16* __restrict__ out_lds, int ldo,
                                           int lane)
{
    const int nm = lane & 15;   // m for A-frag, n for B-frag
    const int hi = lane >> 4;

    v8f acc[4] = {};

#pragma unroll
    for (int k0 = 0; k0 < KTOT; k0 += 32) {
        const _Float16* rb = Wt + (size_t)(n0 + nm) * KTOT + k0 + hi * 8;
        v8h b0 = *(const v8h*)(rb);
        v8h b1 = *(const v8h*)(rb + 16);
        v16h b;
#pragma unroll
        for (int i = 0; i < 8; ++i) { b[i] = b0[i]; b[i + 8] = b1[i]; }
#pragma unroll
        for (int mt = 0; mt < 4; ++mt) {
            const _Float16* ra = A_lds + (size_t)(mt * 16 + nm) * LDA + k0 + hi * 8;
            v8h a0 = *(const v8h*)(ra);
            v8h a1 = *(const v8h*)(ra + 16);
            v16h a;
#pragma unroll
            for (int i = 0; i < 8; ++i) { a[i] = a0[i]; a[i + 8] = a1[i]; }
            acc[mt] = __builtin_amdgcn_wmma_f32_16x16x32_f16(
                false, a, false, b, (short)0, acc[mt], false, false);
        }
    }

    const float bv = bias[n0 + nm];
#pragma unroll
    for (int mt = 0; mt < 4; ++mt) {
#pragma unroll
        for (int r = 0; r < 8; ++r) {
            float v = acc[mt][r] + bv;
            if (RELU) v = fmaxf(v, 0.0f);
            out_lds[(size_t)(mt * 16 + r + 8 * hi) * ldo + n0 + nm] = (_Float16)v;
        }
    }
}

// ---------------------------------------------------------------------------
// Weight prep: f32 -> f16 with transpose to [N][K] (B-fragments contiguous).
// ---------------------------------------------------------------------------
__global__ __launch_bounds__(256) void prep_weights_kernel(
    const float* __restrict__ Win, const float* __restrict__ Wh,
    const float* __restrict__ Wout,
    _Float16* __restrict__ WinT, _Float16* __restrict__ WhT,
    _Float16* __restrict__ WoutT)
{
    int i = blockIdx.x * 256 + threadIdx.x;
    if (i >= 3 * 256 * 128) return;
    int t = i / 32768, r = i % 32768;

    { int n = r / 256, k = r % 256;     // WinT[t][n(128)][k(256)]
      WinT[i] = (_Float16)Win[(size_t)t * 32768 + (size_t)k * 128 + n]; }
    { int li = r / 16384, rr = r % 16384, n = rr / 128, k = rr % 128;
      WhT[i] = (_Float16)Wh[(size_t)t * 32768 + (size_t)li * 16384 + (size_t)k * 128 + n]; }
    { int n = r / 128, k = r % 128;     // WoutT[t][n(256)][k(128)]
      WoutT[i] = (_Float16)Wout[(size_t)t * 32768 + (size_t)k * 256 + n]; }
}

// ---------------------------------------------------------------------------
// Fused main kernel. Block: 256 threads (8 wave32), 64 batch rows.
// LDS ~104 KB: u tile, 2 ping-pong h tiles, 1 per-tower x staging tile,
// small reduction scratch. Only 6 scalars per row (l[3], s[3]) survive the
// towers; softmax + final score computed from those at the end.
// ---------------------------------------------------------------------------
__global__ __launch_bounds__(256) void fused_towers_kernel(
    const int*   __restrict__ user, const int* __restrict__ item,
    const float* __restrict__ su,   const float* __restrict__ ti,
    const _Float16* __restrict__ WinT, const _Float16* __restrict__ WhT,
    const _Float16* __restrict__ WoutT,
    const float* __restrict__ bin_, const float* __restrict__ bh,
    const float* __restrict__ bout, float* __restrict__ out)
{
    __shared__ __align__(128) char smem[105984];
    _Float16* uT   = (_Float16*)(smem);             // [64][LDU]   33792 B
    _Float16* h0   = (_Float16*)(smem + 33792);     // [64][LDH]   17408 B
    _Float16* h1   = (_Float16*)(smem + 51200);     // [64][LDH]   17408 B
    _Float16* xt   = (_Float16*)(smem + 68608);     // [64][LDU]   33792 B
    float*    pls  = (float*)   (smem + 102400);    // [64][4][2]   2048 B
    float*    lrow = (float*)   (smem + 104448);    // [3][64]       768 B
    float*    srow = (float*)   (smem + 105216);    // [3][64]       768 B

    const int tid  = threadIdx.x;
    const int lane = tid & 31;
    const int wave = tid >> 5;
    const int g0   = blockIdx.x * TILE_B;

    const int erow = tid >> 2;          // epilogue: 4 threads per row
    const int eq   = tid & 3;
    const int uid  = user[g0 + erow];
    const int iid  = item[g0 + erow];
    const float* ik = ti + (size_t)uid * D_EMB;   // i_key  = ti_table[user]
    const float* ie = ti + (size_t)iid * D_EMB;   // i_emb  = ti_table[item]

    // ---- gather u_embed rows (f32 -> f16) into padded LDS tile ----
    for (int idx = tid; idx < TILE_B * 64; idx += 256) {
        int row = idx >> 6;
        int c4  = (idx & 63) << 2;
        int ur  = user[g0 + row];
        const float4 v = *(const float4*)(su + (size_t)ur * D_EMB + c4);
        union { _Float16 h[4]; unsigned long long q; } pk;
        pk.h[0] = (_Float16)v.x; pk.h[1] = (_Float16)v.y;
        pk.h[2] = (_Float16)v.z; pk.h[3] = (_Float16)v.w;
        *(unsigned long long*)(uT + (size_t)row * LDU + c4) = pk.q;
    }
    __syncthreads();

    // ---- 3 towers ----
    for (int t = 0; t < 3; ++t) {
        gemm_stage<256, LDU, true>(uT, WinT + (size_t)t * H_DIM * D_EMB,
                                   wave * 16, bin_ + t * H_DIM, h0, LDH, lane);
        __syncthreads();
        gemm_stage<128, LDH, true>(h0, WhT + (size_t)(t * 2 + 0) * H_DIM * H_DIM,
                                   wave * 16, bh + (t * 2 + 0) * H_DIM, h1, LDH, lane);
        __syncthreads();
        gemm_stage<128, LDH, true>(h1, WhT + (size_t)(t * 2 + 1) * H_DIM * H_DIM,
                                   wave * 16, bh + (t * 2 + 1) * H_DIM, h0, LDH, lane);
        __syncthreads();
        // output layer: N=256 -> two 16-wide tiles per wave, no relu
        gemm_stage<128, LDH, false>(h0, WoutT + (size_t)t * D_EMB * H_DIM,
                                    wave * 16,       bout + t * D_EMB, xt, LDU, lane);
        gemm_stage<128, LDH, false>(h0, WoutT + (size_t)t * D_EMB * H_DIM,
                                    wave * 16 + 128, bout + t * D_EMB, xt, LDU, lane);
        __syncthreads();

        // ---- per-tower: l_t = x_t . i_key ; s_t = x_t . i_emb ----
        {
            const _Float16* xr = xt + (size_t)erow * LDU;
            float lp = 0.f, sp = 0.f;
            for (int d = eq * 64; d < eq * 64 + 64; d += 8) {
                v8h xv = *(const v8h*)(xr + d);
                float4 k0v = *(const float4*)(ik + d);
                float4 k1v = *(const float4*)(ik + d + 4);
                float4 e0v = *(const float4*)(ie + d);
                float4 e1v = *(const float4*)(ie + d + 4);
                float x0 = (float)xv[0], x1 = (float)xv[1], x2 = (float)xv[2], x3 = (float)xv[3];
                float x4 = (float)xv[4], x5 = (float)xv[5], x6 = (float)xv[6], x7 = (float)xv[7];
                lp += x0 * k0v.x + x1 * k0v.y + x2 * k0v.z + x3 * k0v.w
                    + x4 * k1v.x + x5 * k1v.y + x6 * k1v.z + x7 * k1v.w;
                sp += x0 * e0v.x + x1 * e0v.y + x2 * e0v.z + x3 * e0v.w
                    + x4 * e1v.x + x5 * e1v.y + x6 * e1v.z + x7 * e1v.w;
            }
            pls[(erow * 4 + eq) * 2 + 0] = lp;
            pls[(erow * 4 + eq) * 2 + 1] = sp;
        }
        __syncthreads();
        if (tid < 128) {
            int row = tid >> 1, which = tid & 1;
            float s = pls[(row * 4 + 0) * 2 + which] + pls[(row * 4 + 1) * 2 + which]
                    + pls[(row * 4 + 2) * 2 + which] + pls[(row * 4 + 3) * 2 + which];
            if (which == 0) lrow[t * 64 + row] = s;
            else            srow[t * 64 + row] = s;
        }
        __syncthreads();
    }

    // ---- softmax over 3 towers + final score ----
    if (tid < TILE_B) {
        float l0 = lrow[tid], l1 = lrow[64 + tid], l2 = lrow[128 + tid];
        float s0 = srow[tid], s1 = srow[64 + tid], s2 = srow[128 + tid];
        float m  = fmaxf(l0, fmaxf(l1, l2));
        float e0 = __expf(l0 - m), e1 = __expf(l1 - m), e2 = __expf(l2 - m);
        out[g0 + tid] = (e0 * s0 + e1 * s1 + e2 * s2) / (e0 + e1 + e2);
    }
}

// ---------------------------------------------------------------------------
extern "C" void kernel_launch(void* const* d_in, const int* in_sizes, int n_in,
                              void* d_out, int out_size, void* d_ws, size_t ws_size,
                              hipStream_t stream) {
    const int*   user = (const int*)d_in[0];
    const int*   item = (const int*)d_in[1];
    const float* su   = (const float*)d_in[2];
    const float* ti   = (const float*)d_in[3];
    const float* Win  = (const float*)d_in[4];
    const float* bin_ = (const float*)d_in[5];
    const float* Wh   = (const float*)d_in[6];
    const float* bh   = (const float*)d_in[7];
    const float* Wout = (const float*)d_in[8];
    const float* bout = (const float*)d_in[9];
    float* out = (float*)d_out;

    _Float16* WinT  = (_Float16*)d_ws;           // 3*128*256 = 98304 f16
    _Float16* WhT   = WinT + 98304;              // 3*2*128*128 = 98304 f16
    _Float16* WoutT = WhT + 98304;               // 3*256*128 = 98304 f16

    const int nelem = 3 * 256 * 128;
    prep_weights_kernel<<<(nelem + 255) / 256, 256, 0, stream>>>(
        Win, Wh, Wout, WinT, WhT, WoutT);

    const int B = in_sizes[0];                   // 65536
    fused_towers_kernel<<<B / TILE_B, 256, 0, stream>>>(
        user, item, su, ti, WinT, WhT, WoutT, bin_, bh, bout, out);
}
